// ValenceConditionedAttention_4552665334333
// MI455X (gfx1250) — matrix operand order repaired and checked by reading, not compiled, not run on hardware
//
#include <hip/hip_runtime.h>

// ---------------------------------------------------------------------------
// ValenceConditionedAttention on MI455X (gfx1250): bf16 WMMA pipeline + TDM.
// B=4, S=1024, D=1024, H=16, hd=64.
// ---------------------------------------------------------------------------

typedef __attribute__((ext_vector_type(16))) __bf16        bf16x16;
typedef __attribute__((ext_vector_type(8)))  __bf16        bf16x8;
typedef __attribute__((ext_vector_type(8)))  float         f32x8;
typedef __attribute__((ext_vector_type(4)))  unsigned int  u32x4;
typedef __attribute__((ext_vector_type(8)))  int           i32x8;
typedef __attribute__((ext_vector_type(4)))  int           i32x4;

union FragAB {
    bf16x16 v;
    bf16x8  h[2];
};

// fp32 -> bf16, round-to-nearest-even
__device__ __forceinline__ unsigned short f2bf(float f) {
    unsigned int u = __float_as_uint(f);
    u += 0x7FFFu + ((u >> 16) & 1u);
    return (unsigned short)(u >> 16);
}

__device__ __forceinline__ f32x8 wmma_bf16(const FragAB& a, const FragAB& b, f32x8 c) {
    return __builtin_amdgcn_wmma_f32_16x16x32_bf16(false, a.v, false, b.v,
                                                   (short)0, c, false, false);
}

// ---------------------------------------------------------------------------
// Tensor Data Mover: 2D bf16 tile (rows x cols elements, row stride in
// elements) from global -> LDS, packed contiguously. D# built per CDNA5 ISA
// (Group0 128b: count/lds_addr/global_addr/type; Group1 256b: data_size,
// tensor dims, tile dims, dim0 stride). Groups 2/3 zero (<=2D tensor).
// This toolchain's builtin is the 6-arg clang-23 form.
// Issue from one wave only; TDM ignores EXEC and is tracked by TENSORcnt.
// ---------------------------------------------------------------------------
__device__ __forceinline__ void tdm_load_tile_bf16(const void* gptr, void* lptr,
                                                   unsigned rows, unsigned cols,
                                                   unsigned stride_elems) {
    const unsigned long long ga  = (unsigned long long)gptr;
    const unsigned           lds = (unsigned)(unsigned long long)lptr;  // LDS byte offset

    u32x4 g0;
    g0[0] = 1u;                                                  // count=1, user desc
    g0[1] = lds;                                                 // lds_addr [63:32]
    g0[2] = (unsigned)ga;                                        // global_addr[31:0]
    g0[3] = (unsigned)((ga >> 32) & 0x1FFFFFFull) | (2u << 30);  // addr[56:32] | type=2

    i32x8 g1;
    g1[0] = (int)(1u << 16);                                     // data_size=1 -> 2 bytes
    g1[1] = (int)((cols & 0xFFFFu) << 16);                       // tensor_dim0 lo16
    g1[2] = (int)(((cols >> 16) & 0xFFFFu) | ((rows & 0xFFFFu) << 16));  // dim0 hi | dim1 lo
    g1[3] = (int)(((rows >> 16) & 0xFFFFu) | ((cols & 0xFFFFu) << 16));  // dim1 hi | tile_dim0
    g1[4] = (int)(rows & 0xFFFFu);                               // tile_dim1 | tile_dim2=0
    g1[5] = (int)stride_elems;                                   // tensor_dim0_stride lo32
    g1[6] = 0;                                                   // stride hi | dim1_stride lo
    g1[7] = 0;

    const i32x4 z4 = {0, 0, 0, 0};
    const i32x8 z8 = {0, 0, 0, 0, 0, 0, 0, 0};
    __builtin_amdgcn_tensor_load_to_lds(g0, g1, z4, z4, z8, 0);
}

// ---------------------------------------------------------------------------
// Generic C[M=4096,N=1024] = A[M,1024] @ W[N,1024]^T + bias, bf16 WMMA.
// MODE 0: out = q  -> bf16 [B,H,S,64]
// MODE 1: out = k  -> bf16 [B,H,S,64]
// MODE 2: out = vT -> bf16 [B,H,64,S]
// MODE 3: out-projection, A is bf16 ctx (staged via TDM), out fp32 [M,N]
// Block tile 32x64 (2x4 waves of 16x16), BK = 64 (2 WMMA per wave per stage).
// ---------------------------------------------------------------------------
template <int MODE>
__global__ __launch_bounds__(256) void gemm_bf16_k(const void* __restrict__ Aptr,
                                                   const float* __restrict__ W,
                                                   const float* __restrict__ bias,
                                                   void* __restrict__ Optr) {
    __shared__ __attribute__((aligned(32))) unsigned short As[32 * 64];  // 4 KB
    __shared__ __attribute__((aligned(32))) unsigned short Bs[64 * 64];  // 8 KB

    const int tid  = threadIdx.x;
    const int wave = tid >> 5;
    const int lane = tid & 31;
    const int l15  = lane & 15;
    const int lh   = lane >> 4;
    const int wr   = wave >> 2;   // 0..1  (M sub-tile)
    const int wc   = wave & 3;    // 0..3  (N sub-tile)
    const int m0   = blockIdx.y * 32;
    const int n0   = blockIdx.x * 64;

    f32x8 acc = {0.f, 0.f, 0.f, 0.f, 0.f, 0.f, 0.f, 0.f};

    for (int k0 = 0; k0 < 1024; k0 += 64) {
        // ---- stage A tile (32 x 64) ----
        if constexpr (MODE == 3) {
            // pure bf16 copy: Tensor Data Mover, wave 0 issues one 2D tile DMA
            if (wave == 0) {
                tdm_load_tile_bf16((const unsigned short*)Aptr + (size_t)m0 * 1024 + k0,
                                   &As[0], 32u, 64u, 1024u);
            }
        } else {
            for (int i = tid; i < 32 * 64; i += 256) {
                const int r = i >> 6, c = i & 63;
                As[i] = f2bf(((const float*)Aptr)[(size_t)(m0 + r) * 1024 + k0 + c]);
            }
        }
        // ---- stage W tile (64 rows x 64 K), fp32 -> bf16 ----
        for (int i = tid; i < 64 * 64; i += 256) {
            const int r = i >> 6, c = i & 63;
            Bs[i] = f2bf(W[(size_t)(n0 + r) * 1024 + k0 + c]);
        }
        // prefetch next K-step of W into cache while WMMAs run
        if (k0 + 64 < 1024) {
            const int pr = tid & 63;
            __builtin_prefetch(&W[(size_t)(n0 + pr) * 1024 + k0 + 64 + ((tid >> 6) << 4)], 0, 1);
        }
        if constexpr (MODE == 3) {
            if (wave == 0) __builtin_amdgcn_s_wait_tensorcnt(0);
        }
        __syncthreads();

        for (int kk = 0; kk < 2; ++kk) {
            FragAB fa, fb;
            const int akb = kk * 32 + lh * 8;  // A: lanes0-15 K{0-7,16-23}, 16-31 K{8-15,24-31}
            const int arow = wr * 16 + l15;
            fa.h[0] = *(const bf16x8*)&As[arow * 64 + akb];
            fa.h[1] = *(const bf16x8*)&As[arow * 64 + akb + 16];
            const int brow = wc * 16 + l15;
            fb.h[0] = *(const bf16x8*)&Bs[brow * 64 + akb];  // B: same K order as A
            fb.h[1] = *(const bf16x8*)&Bs[brow * 64 + akb + 16];
            acc = wmma_bf16(fa, fb, acc);
        }
        __syncthreads();
    }

    // Epilogue: C/D layout lane=N (l15), VGPR r -> M = r + 8*lh
    const int   n  = n0 + wc * 16 + l15;
    const float bv = bias[n];
    for (int r = 0; r < 8; ++r) {
        const int   m   = m0 + wr * 16 + r + 8 * lh;
        const float val = acc[r] + bv;
        if constexpr (MODE == 3) {
            ((float*)Optr)[(size_t)m * 1024 + n] = val;
        } else {
            const int bb = m >> 10, s = m & 1023;
            const int hh = n >> 6, d = n & 63;
            unsigned short* O = (unsigned short*)Optr;
            if constexpr (MODE == 2) {
                O[(((size_t)bb * 16 + hh) * 64 + d) * 1024 + s] = f2bf(val);  // v transposed
            } else {
                O[(((size_t)bb * 16 + hh) * 1024 + s) * 64 + d] = f2bf(val);  // q,k
            }
        }
    }
}

// ---------------------------------------------------------------------------
// Fused attention per (b, h, 16-query-row tile).
// Q tile staged via TDM; scores(16x1024 f32) in LDS -> softmax ->
// atomic attn_avg += attn*scale/H -> ctx(16x64) via WMMA against vT.
// LDS: 2K (Q) + 64K (scores) + 32K (attn bf16) + 8K (ctx partials) = 106KB.
// ---------------------------------------------------------------------------
__global__ __launch_bounds__(256) void attn_fused_k(const unsigned short* __restrict__ qws,
                                                    const unsigned short* __restrict__ kws,
                                                    const unsigned short* __restrict__ vtws,
                                                    const float* __restrict__ valence,
                                                    unsigned short* __restrict__ ctxws,
                                                    float* __restrict__ attn_avg) {
    __shared__ __attribute__((aligned(32))) unsigned short sQ[16 * 64];
    __shared__ __attribute__((aligned(32))) float          sS[16 * 1024];
    __shared__ __attribute__((aligned(32))) unsigned short sA[16 * 1024];
    __shared__ __attribute__((aligned(32))) float          sC[2 * 16 * 64];

    const int tid  = threadIdx.x;
    const int wave = tid >> 5;
    const int lane = tid & 31;
    const int l15  = lane & 15;
    const int lh   = lane >> 4;
    const int bh   = blockIdx.y;          // b*16 + h
    const int b    = bh >> 4;
    const int h    = bh & 15;
    const int q0   = blockIdx.x * 16;

    const float vmean = 0.25f * (valence[0] + valence[1] + valence[2] + valence[3]);
    const float sc    = (0.5f + 1.5f * vmean) * (1.0f / 16.0f);  // scale / H

    // Q tile (16x64 bf16, contiguous rows of 64): one TDM 2D tile DMA
    if (wave == 0) {
        tdm_load_tile_bf16(qws + ((size_t)bh * 1024 + q0) * 64, &sQ[0], 16u, 64u, 64u);
        __builtin_amdgcn_s_wait_tensorcnt(0);
    }
    __syncthreads();

    // Phase A: scores = Q K^T / 8.  Wave w owns 8 column tiles (128 cols).
    const size_t kbase_g = (size_t)bh * 1024 * 64;
    for (int ct = wave * 8; ct < wave * 8 + 8; ++ct) {
        f32x8 acc = {0.f, 0.f, 0.f, 0.f, 0.f, 0.f, 0.f, 0.f};
        for (int kk = 0; kk < 2; ++kk) {
            FragAB fa, fb;
            const int akb = kk * 32 + lh * 8;
            fa.h[0] = *(const bf16x8*)&sQ[l15 * 64 + akb];
            fa.h[1] = *(const bf16x8*)&sQ[l15 * 64 + akb + 16];
            const int krow = ct * 16 + l15;
            fb.h[0] = *(const bf16x8*)&kws[kbase_g + (size_t)krow * 64 + akb];
            fb.h[1] = *(const bf16x8*)&kws[kbase_g + (size_t)krow * 64 + akb + 16];
            acc = wmma_bf16(fa, fb, acc);
        }
        const int n = ct * 16 + l15;
        for (int r = 0; r < 8; ++r) sS[(r + 8 * lh) * 1024 + n] = acc[r] * 0.125f;
    }
    __syncthreads();

    // Phase B: softmax over each of 16 rows (16 threads per row), sC as scratch.
    float*    red = sC;
    const int row = tid >> 4;
    const int sub = tid & 15;
    float     lmax = -3.0e38f;
    for (int j = 0; j < 64; ++j) lmax = fmaxf(lmax, sS[row * 1024 + sub + j * 16]);
    red[row * 16 + sub] = lmax;
    __syncthreads();
    float mx = red[row * 16];
    for (int j = 1; j < 16; ++j) mx = fmaxf(mx, red[row * 16 + j]);
    __syncthreads();
    float lsum = 0.f;
    for (int j = 0; j < 64; ++j) {
        const int   idx = row * 1024 + sub + j * 16;
        const float e   = __expf(sS[idx] - mx);
        sS[idx]         = e;
        lsum += e;
    }
    red[row * 16 + sub] = lsum;
    __syncthreads();
    float ssum = 0.f;
    for (int j = 0; j < 16; ++j) ssum += red[row * 16 + j];
    const float inv = 1.0f / ssum;
    for (int j = 0; j < 64; ++j) {
        const int   col = sub + j * 16;
        const float v   = sS[row * 1024 + col] * inv;
        sA[row * 1024 + col] = f2bf(v);
        atomicAdd(&attn_avg[((size_t)b * 1024 + q0 + row) * 1024 + col], v * sc);
    }
    __syncthreads();

    // Phase C: ctx = attn @ V.  Wave (nt, kh): nt = d-tile 0..3, kh = K half.
    const int    nt = wave & 3;
    const int    kh = wave >> 2;
    f32x8        acc = {0.f, 0.f, 0.f, 0.f, 0.f, 0.f, 0.f, 0.f};
    const size_t vbase = (size_t)bh * 64 * 1024;
    for (int ks = 0; ks < 16; ++ks) {
        const int kstep = kh * 16 + ks;
        FragAB    fa, fb;
        const int kb = kstep * 32 + lh * 8;
        fa.h[0] = *(const bf16x8*)&sA[l15 * 1024 + kb];
        fa.h[1] = *(const bf16x8*)&sA[l15 * 1024 + kb + 16];
        const int d = nt * 16 + l15;
        fb.h[0] = *(const bf16x8*)&vtws[vbase + (size_t)d * 1024 + kb];
        fb.h[1] = *(const bf16x8*)&vtws[vbase + (size_t)d * 1024 + kb + 16];
        acc = wmma_bf16(fa, fb, acc);
    }
    for (int r = 0; r < 8; ++r)
        sC[kh * 1024 + (r + 8 * lh) * 64 + nt * 16 + l15] = acc[r];
    __syncthreads();
    for (int i = tid; i < 16 * 64; i += 256) {
        const int   m = i >> 6, d = i & 63;
        const float v = sC[i] + sC[1024 + i];
        ctxws[((size_t)b * 1024 + q0 + m) * 1024 + h * 64 + d] = f2bf(v);
    }
}

// ---------------------------------------------------------------------------
// LayerNorm over last dim (1024), one block per row.
// ---------------------------------------------------------------------------
__global__ __launch_bounds__(256) void layernorm_k(const float* __restrict__ x,
                                                   const float* __restrict__ g,
                                                   const float* __restrict__ bta,
                                                   float* __restrict__ out) {
    __shared__ float red[256];
    const int    row = blockIdx.x;
    const int    tid = threadIdx.x;
    const float* xr  = x + (size_t)row * 1024;
    float        v[4], s = 0.f, s2 = 0.f;
    for (int j = 0; j < 4; ++j) {
        v[j] = xr[tid + j * 256];
        s += v[j];
        s2 += v[j] * v[j];
    }
    red[tid] = s;
    __syncthreads();
    for (int off = 128; off; off >>= 1) {
        if (tid < off) red[tid] += red[tid + off];
        __syncthreads();
    }
    const float mean = red[0] * (1.0f / 1024.0f);
    __syncthreads();
    red[tid] = s2;
    __syncthreads();
    for (int off = 128; off; off >>= 1) {
        if (tid < off) red[tid] += red[tid + off];
        __syncthreads();
    }
    const float var  = red[0] * (1.0f / 1024.0f) - mean * mean;
    const float rstd = rsqrtf(var + 1e-5f);
    for (int j = 0; j < 4; ++j) {
        const int c = tid + j * 256;
        out[(size_t)row * 1024 + c] = (v[j] - mean) * rstd * g[c] + bta[c];
    }
}

__global__ __launch_bounds__(256) void zero_k(float* __restrict__ p, size_t n) {
    const size_t i = (size_t)blockIdx.x * blockDim.x + threadIdx.x;
    if (i < n) p[i] = 0.f;
}

// ---------------------------------------------------------------------------
extern "C" void kernel_launch(void* const* d_in, const int* in_sizes, int n_in,
                              void* d_out, int out_size, void* d_ws, size_t ws_size,
                              hipStream_t stream) {
    const float* query   = (const float*)d_in[0];
    const float* key_    = (const float*)d_in[1];
    const float* value   = (const float*)d_in[2];
    const float* valence = (const float*)d_in[3];
    const float* ipw     = (const float*)d_in[4];
    const float* ipb     = (const float*)d_in[5];
    const float* outw    = (const float*)d_in[6];
    const float* outb    = (const float*)d_in[7];
    const float* lng     = (const float*)d_in[8];
    const float* lnb     = (const float*)d_in[9];

    float* out      = (float*)d_out;                        // normed [4,1024,1024]
    float* attn_avg = out + (size_t)4 * 1024 * 1024;        // [4,1024,1024]

    char*           ws    = (char*)d_ws;                    // ~48 MB used
    unsigned short* qws   = (unsigned short*)(ws);                      // 8 MB
    unsigned short* kws   = (unsigned short*)(ws + ((size_t)8 << 20));  // 8 MB
    unsigned short* vtws  = (unsigned short*)(ws + ((size_t)16 << 20)); // 8 MB
    unsigned short* ctxws = (unsigned short*)(ws + ((size_t)24 << 20)); // 8 MB
    float*          aout  = (float*)(ws + ((size_t)32 << 20));          // 16 MB

    const size_t navg = (size_t)4 * 1024 * 1024;
    zero_k<<<16384, 256, 0, stream>>>(attn_avg, navg);

    const dim3 gg(16, 128);  // N/64, M/32
    gemm_bf16_k<0><<<gg, 256, 0, stream>>>(query, ipw, ipb, qws);
    gemm_bf16_k<1><<<gg, 256, 0, stream>>>(key_, ipw + (size_t)1024 * 1024, ipb + 1024, kws);
    gemm_bf16_k<2><<<gg, 256, 0, stream>>>(value, ipw + (size_t)2 * 1024 * 1024, ipb + 2048, vtws);

    attn_fused_k<<<dim3(64, 64), 256, 0, stream>>>(qws, kws, vtws, valence, ctxws, attn_avg);

    gemm_bf16_k<3><<<gg, 256, 0, stream>>>(ctxws, outw, outb, aout);

    layernorm_k<<<4096, 256, 0, stream>>>(aout, lng, lnb, out);
}
